// MultimodalDotProductAttention_10866267258968
// MI455X (gfx1250) — matrix operand order repaired and checked by reading, not compiled, not run on hardware
//
#include <hip/hip_runtime.h>
#include <hip/hip_bf16.h>
#include <math.h>

typedef __attribute__((ext_vector_type(16))) _Float16 v16h;
typedef __attribute__((ext_vector_type(8)))  float    v8f;

#define DIM     256
#define HEADS   8
#define HD      32
#define S_TOTAL 4608      // 1536+1536+768+768 per batch
#define S_MAIN  3072
#define NBATCH  2
#define QSCALE  0.17677669529663687f  // 1/sqrt(32)

// sched_group_barrier masks
#define SG_WMMA 0x008
#define SG_VMRD 0x020

// Pipeline directive: 4 vmem reads, then (1 wmma, 2 vmem reads) x7, then 1 wmma.
// Keeps one 32-K B-chunk in flight behind every WMMA instead of wait-loadcnt-0.
__device__ __forceinline__ void sched_pipeline_gemm8() {
  __builtin_amdgcn_sched_group_barrier(SG_VMRD, 4, 0);
#pragma unroll
  for (int i = 0; i < 7; ++i) {
    __builtin_amdgcn_sched_group_barrier(SG_WMMA, 1, 0);
    __builtin_amdgcn_sched_group_barrier(SG_VMRD, 2, 0);
  }
  __builtin_amdgcn_sched_group_barrier(SG_WMMA, 1, 0);
}

// ---------------------------------------------------------------------------
// K0: transpose + convert weights to f16, column(n)-major: Wt[m][n][k]
// ---------------------------------------------------------------------------
__global__ void k_transpose_w(const float* __restrict__ W, _Float16* __restrict__ Wt,
                              int K, int N, int total) {
  int idx = blockIdx.x * blockDim.x + threadIdx.x;
  if (idx >= total) return;
  int m   = idx / (K * N);
  int rem = idx % (K * N);
  int k = rem / N, n = rem % N;
  Wt[((size_t)m * N + n) * K + k] = (_Float16)W[idx];
}

// ---------------------------------------------------------------------------
// K1: QKV projection GEMM (f16 WMMA, f32 acc) + scatter to Q / K / V^T
// grid: 2 batches * 288 row-tiles; block = 128 thr = 4 waves
// ---------------------------------------------------------------------------
__global__ __launch_bounds__(128)
void k_qkv(const float* __restrict__ x0, const float* __restrict__ x1,
           const float* __restrict__ x2, const float* __restrict__ x3,
           const float* __restrict__ bqkv, const _Float16* __restrict__ Wqkvt,
           _Float16* __restrict__ Qb, _Float16* __restrict__ Kb,
           _Float16* __restrict__ Vt) {
  const int wid = threadIdx.x >> 5, lane = threadIdx.x & 31;
  const int blk = blockIdx.x;
  const int b = blk / 288, rt = blk % 288;
  const int q0 = rt * 16;

  int mod, lr, smod; const float* xp;
  if (q0 < 1536)      { mod = 0; xp = x0; lr = q0;        smod = 1536; }
  else if (q0 < 3072) { mod = 1; xp = x1; lr = q0 - 1536; smod = 1536; }
  else if (q0 < 3840) { mod = 2; xp = x2; lr = q0 - 3072; smod = 768;  }
  else                { mod = 3; xp = x3; lr = q0 - 3840; smod = 768;  }

  const int rr   = lane & 15;
  const int base = (lane < 16) ? 0 : 16;   // A-layout K-chunk base
  const int hb   = (lane < 16) ? 0 : 16;   // B-layout K split
  const int rowb = (lane < 16) ? 0 : 8;    // D-layout M split
  const int cc   = lane & 15;

  // Hoist A-tile: x row in f32 -> 8 v16h chunks (K = 0..255 in steps of 32)
  const float* xrow = xp + ((size_t)(b * smod + lr + rr)) * DIM;
  v16h A[8];
#pragma unroll
  for (int kc = 0; kc < 8; ++kc) {
    const int k0 = kc * 32;
    float fl[16];
    *(float4*)&fl[0]  = *(const float4*)(xrow + k0 + base);
    *(float4*)&fl[4]  = *(const float4*)(xrow + k0 + base + 4);
    *(float4*)&fl[8]  = *(const float4*)(xrow + k0 + base + 8);
    *(float4*)&fl[12] = *(const float4*)(xrow + k0 + base + 12);
    _Float16* ah = (_Float16*)&A[kc];
#pragma unroll
    for (int i = 0; i < 16; ++i) ah[i] = (_Float16)fl[i];
  }

  for (int nt = wid; nt < 48; nt += 4) {         // 768/16 = 48 N-tiles
    const int n = nt * 16 + cc;
    const _Float16* wp = Wqkvt + ((size_t)(mod * 768 + n)) * DIM + hb;
    v16h B[8];
#pragma unroll
    for (int kc = 0; kc < 8; ++kc) {
      ((uint4*)&B[kc])[0] = *(const uint4*)(wp + kc * 32);
      ((uint4*)&B[kc])[1] = *(const uint4*)(wp + kc * 32 + 8);
    }
    v8f acc = {};
#pragma unroll
    for (int kc = 0; kc < 8; ++kc)
      acc = __builtin_amdgcn_wmma_f32_16x16x32_f16(false, A[kc], false, B[kc],
                                                   (short)0, acc, false, false);
    sched_pipeline_gemm8();
    // ---- wave-uniform tile coordinates (keeps the scatter branch scalar)
    const int which = nt >> 4;                 // (nt*16) / 256
    const int head  = (nt >> 1) & 7;           // ((nt*16) / 32) % 8
    const int d     = ((nt & 1) << 4) + cc;    // (nt*16) % 32 + cc
    const float bias = bqkv[mod * 768 + n];
    const size_t bh = (size_t)(b * HEADS + head);
#pragma unroll
    for (int v = 0; v < 8; ++v) {
      const float val = acc[v] + bias;
      const int grow = q0 + v + rowb;
      if (which == 0)
        Qb[(bh * S_TOTAL + grow) * HD + d] = (_Float16)(val * QSCALE);
      else if (which == 1)
        Kb[(bh * S_TOTAL + grow) * HD + d] = (_Float16)val;
      else
        Vt[(bh * HD + d) * S_TOTAL + grow] = (_Float16)val;
    }
  }
}

// ---------------------------------------------------------------------------
// K2: flash attention, one wave per (b, head, 16-query tile)
// grid: 4608/4 = 1152 blocks of 128 thr
// ---------------------------------------------------------------------------
__global__ __launch_bounds__(128)
void k_attn(const _Float16* __restrict__ Qb, const _Float16* __restrict__ Kb,
            const _Float16* __restrict__ Vt, _Float16* __restrict__ attn) {
  __shared__ _Float16 lds[4][16][32];            // per-wave P bounce tile
  const int wid = threadIdx.x >> 5, lane = threadIdx.x & 31;
  const int t = blockIdx.x * 4 + wid;            // 0..4607
  const int b  = t / (HEADS * 288);
  const int r  = t % (HEADS * 288);
  const int h  = r / 288;
  const int q0 = (r % 288) * 16;
  const size_t bh = (size_t)(b * HEADS + h);

  const _Float16* Kbase = Kb + bh * (size_t)S_TOTAL * HD;
  const _Float16* Vbase = Vt + bh * (size_t)HD * S_TOTAL;

  const int rr   = lane & 15;
  const int base = (lane < 16) ? 0 : 16;
  const int hb   = (lane < 16) ? 0 : 16;
  const int rowb = (lane < 16) ? 0 : 8;
  const int cc   = lane & 15;

  // Q tile (already scaled by 1/sqrt(d)) in A layout
  const _Float16* qrow = Qb + (bh * S_TOTAL + q0 + rr) * HD;
  v16h qa;
  ((uint4*)&qa)[0] = *(const uint4*)(qrow + base);
  ((uint4*)&qa)[1] = *(const uint4*)(qrow + base + 8);

  v8f o1 = {}, o2 = {};
  float mrow[8], lrow[8];
#pragma unroll
  for (int v = 0; v < 8; ++v) { mrow[v] = -3.0e38f; lrow[v] = 0.f; }

  int start2, len2;
  if (q0 < 3072)      { start2 = 0;    len2 = 0;   }   // main: keys [0,3072)
  else if (q0 < 3840) { start2 = 3072; len2 = 768; }   // aux0: + own keys
  else                { start2 = 3840; len2 = 768; }   // aux1: + own keys

  _Float16* pl = &lds[wid][0][0];

  for (int pass = 0; pass < 2; ++pass) {
    const int ks = (pass == 0) ? 0 : start2;
    const int kl = (pass == 0) ? S_MAIN : len2;
    for (int kb = ks; kb < ks + kl; kb += 32) {
      // ---- issue ALL K and V tile loads up front (one clause); V arrives
      //      while the score WMMAs + softmax VALU work execute
      const _Float16* kp = Kbase + (size_t)(kb + cc) * HD + hb;
      const _Float16* vp = Vbase + (size_t)cc * S_TOTAL + kb + hb;
      v16h b1, b2, vb1, vb2;
      ((uint4*)&b1)[0]  = *(const uint4*)kp;
      ((uint4*)&b1)[1]  = *(const uint4*)(kp + 8);
      ((uint4*)&b2)[0]  = *(const uint4*)(kp + 16 * HD);
      ((uint4*)&b2)[1]  = *(const uint4*)(kp + 16 * HD + 8);
      ((uint4*)&vb1)[0] = *(const uint4*)vp;
      ((uint4*)&vb1)[1] = *(const uint4*)(vp + 8);
      ((uint4*)&vb2)[0] = *(const uint4*)(vp + 16 * S_TOTAL);
      ((uint4*)&vb2)[1] = *(const uint4*)(vp + 16 * S_TOTAL + 8);
      __builtin_prefetch((const void*)(kp + 32 * HD), 0, 1);
      __builtin_prefetch((const void*)(vp + 32), 0, 1);

      // ---- scores: Q(16x32) x K^T(32x32) -> two 16x16 f32 tiles
      v8f z = {};
      v8f s1 = __builtin_amdgcn_wmma_f32_16x16x32_f16(false, qa, false, b1,
                                                      (short)0, z, false, false);
      v8f s2 = __builtin_amdgcn_wmma_f32_16x16x32_f16(false, qa, false, b2,
                                                      (short)0, z, false, false);
      // make the scheduler keep all 8 tile loads ahead of the score WMMAs
      __builtin_amdgcn_sched_group_barrier(SG_VMRD, 8, 0);
      __builtin_amdgcn_sched_group_barrier(SG_WMMA, 2, 0);

      // ---- online softmax (row stats via 16-lane xor reductions)
      float sc[8];
#pragma unroll
      for (int v = 0; v < 8; ++v) {
        float bm = fmaxf(s1[v], s2[v]);
        bm = fmaxf(bm, __shfl_xor(bm, 1));
        bm = fmaxf(bm, __shfl_xor(bm, 2));
        bm = fmaxf(bm, __shfl_xor(bm, 4));
        bm = fmaxf(bm, __shfl_xor(bm, 8));
        const float mn = fmaxf(mrow[v], bm);
        const float s  = __expf(mrow[v] - mn);
        const float p1 = __expf(s1[v] - mn);
        const float p2 = __expf(s2[v] - mn);
        float rs = p1 + p2;
        rs += __shfl_xor(rs, 1); rs += __shfl_xor(rs, 2);
        rs += __shfl_xor(rs, 4); rs += __shfl_xor(rs, 8);
        lrow[v] = lrow[v] * s + rs;
        mrow[v] = mn;
        sc[v] = s;
        pl[(v + rowb) * 32 + cc]      = (_Float16)p1;   // D-layout -> row major
        pl[(v + rowb) * 32 + cc + 16] = (_Float16)p2;
      }
      asm volatile("s_wait_dscnt 0" ::: "memory");      // CDNA5 split DS counter
      // ---- reload P in A layout
      v16h pa;
      ((uint4*)&pa)[0] = *(const uint4*)&lds[wid][rr][base];
      ((uint4*)&pa)[1] = *(const uint4*)&lds[wid][rr][base + 8];
#pragma unroll
      for (int v = 0; v < 8; ++v) { o1[v] *= sc[v]; o2[v] *= sc[v]; }
      // ---- O += P(16x32) x V(32x32); V tiles already resident in VGPRs
      o1 = __builtin_amdgcn_wmma_f32_16x16x32_f16(false, pa, false, vb1,
                                                  (short)0, o1, false, false);
      o2 = __builtin_amdgcn_wmma_f32_16x16x32_f16(false, pa, false, vb2,
                                                  (short)0, o2, false, false);
    }
  }

  // ---- normalize and store merged-head f16 output
  _Float16* op = attn + ((size_t)b * S_TOTAL + q0) * DIM + h * HD;
#pragma unroll
  for (int v = 0; v < 8; ++v) {
    const float inv = 1.0f / lrow[v];
    const int grow = v + rowb;
    op[(size_t)grow * DIM + cc]      = (_Float16)(o1[v] * inv);
    op[(size_t)grow * DIM + cc + 16] = (_Float16)(o2[v] * inv);
  }
}

// ---------------------------------------------------------------------------
// K3: per-modality output projection, f32 result to d_out
// ---------------------------------------------------------------------------
__global__ __launch_bounds__(128)
void k_proj(const _Float16* __restrict__ attn, const _Float16* __restrict__ Wprojt,
            const float* __restrict__ bproj, float* __restrict__ out) {
  const int wid = threadIdx.x >> 5, lane = threadIdx.x & 31;
  const int blk = blockIdx.x;
  const int b = blk / 288, rt = blk % 288;
  const int q0 = rt * 16;
  const int mod = (q0 < 1536) ? 0 : (q0 < 3072) ? 1 : (q0 < 3840) ? 2 : 3;

  const int rr   = lane & 15;
  const int base = (lane < 16) ? 0 : 16;
  const int hb   = (lane < 16) ? 0 : 16;
  const int rowb = (lane < 16) ? 0 : 8;
  const int cc   = lane & 15;

  const _Float16* arow = attn + ((size_t)(b * S_TOTAL) + q0 + rr) * DIM;
  v16h A[8];
#pragma unroll
  for (int kc = 0; kc < 8; ++kc) {
    ((uint4*)&A[kc])[0] = *(const uint4*)(arow + kc * 32 + base);
    ((uint4*)&A[kc])[1] = *(const uint4*)(arow + kc * 32 + base + 8);
  }

  for (int nt = wid; nt < 16; nt += 4) {         // 256/16 N-tiles
    const int n = nt * 16 + cc;
    const _Float16* wp = Wprojt + ((size_t)(mod * 256 + n)) * DIM + hb;
    v16h B[8];
#pragma unroll
    for (int kc = 0; kc < 8; ++kc) {
      ((uint4*)&B[kc])[0] = *(const uint4*)(wp + kc * 32);
      ((uint4*)&B[kc])[1] = *(const uint4*)(wp + kc * 32 + 8);
    }
    v8f acc = {};
#pragma unroll
    for (int kc = 0; kc < 8; ++kc)
      acc = __builtin_amdgcn_wmma_f32_16x16x32_f16(false, A[kc], false, B[kc],
                                                   (short)0, acc, false, false);
    sched_pipeline_gemm8();
    const float bias = bproj[mod * 256 + n];
    float* orow = out + ((size_t)(b * S_TOTAL) + q0) * DIM;
#pragma unroll
    for (int v = 0; v < 8; ++v)
      orow[(size_t)(v + rowb) * DIM + n] = acc[v] + bias;
  }
}

// ---------------------------------------------------------------------------
extern "C" void kernel_launch(void* const* d_in, const int* in_sizes, int n_in,
                              void* d_out, int out_size, void* d_ws, size_t ws_size,
                              hipStream_t stream) {
  const float* x0    = (const float*)d_in[0];
  const float* x1    = (const float*)d_in[1];
  const float* x2    = (const float*)d_in[2];
  const float* x3    = (const float*)d_in[3];
  const float* Wqkv  = (const float*)d_in[4];
  const float* bqkv  = (const float*)d_in[5];
  const float* Wproj = (const float*)d_in[6];
  const float* bproj = (const float*)d_in[7];
  float* out = (float*)d_out;

  // workspace carve-out (f16), all sizes 256B-aligned
  char* ws = (char*)d_ws;
  _Float16* Wqkvt  = (_Float16*)ws; ws += (size_t)4 * 768 * 256 * 2;            // 1.5 MB
  _Float16* Wprojt = (_Float16*)ws; ws += (size_t)4 * 256 * 256 * 2;            // 0.5 MB
  _Float16* Qb     = (_Float16*)ws; ws += (size_t)NBATCH * HEADS * S_TOTAL * HD * 2; // 4.5 MB
  _Float16* Kb     = (_Float16*)ws; ws += (size_t)NBATCH * HEADS * S_TOTAL * HD * 2;
  _Float16* Vt     = (_Float16*)ws; ws += (size_t)NBATCH * HEADS * HD * S_TOTAL * 2;
  _Float16* attnb  = (_Float16*)ws; ws += (size_t)NBATCH * S_TOTAL * DIM * 2;

  {
    int tot = 4 * 256 * 768;
    k_transpose_w<<<(tot + 255) / 256, 256, 0, stream>>>(Wqkv, Wqkvt, 256, 768, tot);
  }
  {
    int tot = 4 * 256 * 256;
    k_transpose_w<<<(tot + 255) / 256, 256, 0, stream>>>(Wproj, Wprojt, 256, 256, tot);
  }
  k_qkv <<<NBATCH * 288, 128, 0, stream>>>(x0, x1, x2, x3, bqkv, Wqkvt, Qb, Kb, Vt);
  k_attn<<<(NBATCH * HEADS * 288) / 4, 128, 0, stream>>>(Qb, Kb, Vt, attnb);
  k_proj<<<NBATCH * 288, 128, 0, stream>>>(attnb, Wprojt, bproj, out);
}
